// AttentionCell_11158325035105
// MI455X (gfx1250) — compile-verified, matrix operand order checked by reading, not run on hardware
//
#include <hip/hip_runtime.h>
#include <cstdint>
#include <cstddef>

// ---------------------------------------------------------------------------
// Types for CDNA5 WMMA (wave32): v_wmma_f32_16x16x32_bf16
// ---------------------------------------------------------------------------
typedef __attribute__((ext_vector_type(16))) __bf16       v16bf;
typedef __attribute__((ext_vector_type(8)))  float        v8f;
typedef __attribute__((ext_vector_type(4)))  unsigned int uv4;

union Frag {
    uv4   u[2];   // 8 dwords = 16 bf16
    v16bf v;
};

__device__ __forceinline__ unsigned short f2bf(float x) {
    // round-to-nearest-even f32 -> bf16 (bit pattern)
    unsigned int u = __float_as_uint(x);
    u += 0x7FFFu + ((u >> 16) & 1u);
    return (unsigned short)(u >> 16);
}

__device__ __forceinline__ float sigmoidf_(float x) {
    return 1.0f / (1.0f + expf(-x));
}

// ---------------------------------------------------------------------------
// Problem dims
// ---------------------------------------------------------------------------
static constexpr int B_  = 256;
static constexpr int T_  = 256;
static constexpr int DE_ = 512;   // D_ENC
static constexpr int H_  = 512;
static constexpr int NC_ = 96;    // NUM_CLASSES
static constexpr int KX_ = DE_ + NC_ + H_;    // 1120 = 35*32, x_full K
static constexpr int N4H_ = 4 * H_;           // 2048

// ---------------------------------------------------------------------------
// Kernel 1: pack a row-major f32 weight W[K][N] into bf16 WMMA B-fragments.
// Fragment (nb, kc) = 256 dwords; lane holds dwords [lane*8 .. lane*8+7].
// Lane l: n = nb*16 + (l&15), half = l>>4.
// Dword j holds bf16 pair for K = kc*32 + koff(j,half), koff(j,half)+1 where
//   koff = (j<4 ? 2*j : 16 + 2*(j-4)) + half*8   (CDNA5 16-bit operand layout)
// ---------------------------------------------------------------------------
__global__ void swizzle_b_k(const float* __restrict__ W, unsigned int* __restrict__ dst,
                            int N, int KClocal, int kcBase, int KCtot)
{
    int tid   = blockIdx.x * 256 + threadIdx.x;
    int total = (N / 16) * KClocal * 256;
    if (tid >= total) return;
    int per_nb = KClocal * 256;
    int nb   = tid / per_nb;
    int rem  = tid - nb * per_nb;
    int kcl  = rem >> 8;
    int r2   = rem & 255;
    int lane = r2 >> 3;
    int j    = r2 & 7;
    int half = lane >> 4;
    int n    = nb * 16 + (lane & 15);
    int koff = ((j < 4) ? (2 * j) : (16 + 2 * (j - 4))) + half * 8;
    int kk   = kcl * 32 + koff;
    unsigned int lo = f2bf(W[(size_t)kk * N + n]);
    unsigned int hi = f2bf(W[(size_t)(kk + 1) * N + n]);
    dst[(((size_t)nb * KCtot + kcBase + kcl) << 8) + lane * 8 + j] = lo | (hi << 16);
}

// ---------------------------------------------------------------------------
// Kernel 2: generic bf16 WMMA GEMM,  C[M,N] = A[M,K](f32) x Bfrag + epilogue
//   block = 128 threads (4 waves). WG tile: 64 rows x 64 cols.
//   wave w handles n-block nbg = blockIdx.y*4 + w (16 cols), 4 M-tiles.
//   A staged in LDS (f32 -> bf16) in 64x256 chunks (handles K=1120 tail of 96).
// mode 0: C[row*N+col] = acc + bias[col]
// mode 1: attention score epilogue:
//         t = tanh(acc + hproj[row>>8][col]) * wscore[col]
//         reduce over 16 lanes + 4 waves -> epart[blockIdx.y*M + row]
// ---------------------------------------------------------------------------
__global__ __launch_bounds__(128)
void gemm_wmma_k(const float* __restrict__ A, const unsigned int* __restrict__ Bfrag,
                 float* __restrict__ C, const float* __restrict__ bias,
                 int M, int N, int K, int KCtot, int mode,
                 const float* __restrict__ hproj, const float* __restrict__ wscore,
                 float* __restrict__ epart)
{
    __shared__ __attribute__((aligned(16))) unsigned short As[64 * 256];
    __shared__ float eacc[4 * 64];

    const int tid  = threadIdx.x;
    const int w    = tid >> 5;
    const int lane = tid & 31;
    const int half = lane >> 4;
    const int l16  = lane & 15;
    const int rowbase = blockIdx.x * 64;
    const int nbg  = blockIdx.y * 4 + w;
    const int col  = nbg * 16 + l16;

    v8f acc[4];
#pragma unroll
    for (int m = 0; m < 4; ++m)
#pragma unroll
        for (int v = 0; v < 8; ++v) acc[m][v] = 0.0f;

    for (int k0 = 0; k0 < K; k0 += 256) {
        const int cc = (K - k0 < 256) ? (K - k0) : 256;
        __syncthreads();
        const int total = 64 * cc;
        if (cc == 256) {
            for (int i = tid; i < total; i += 128) {
                int r = i >> 8, c = i & 255;
                As[(r << 8) + c] = f2bf(A[(size_t)(rowbase + r) * K + k0 + c]);
            }
        } else {
            for (int i = tid; i < total; i += 128) {
                int r = i / cc, c = i - r * cc;
                As[(r << 8) + c] = f2bf(A[(size_t)(rowbase + r) * K + k0 + c]);
            }
        }
        __syncthreads();
        const int nkc  = cc >> 5;
        const int kcg0 = k0 >> 5;
        for (int kc = 0; kc < nkc; ++kc) {
            Frag bf;
            const unsigned int* bp =
                Bfrag + (((size_t)nbg * KCtot + kcg0 + kc) << 8) + lane * 8;
            bf.u[0] = *(const uv4*)(bp);
            bf.u[1] = *(const uv4*)(bp + 4);
#pragma unroll
            for (int m = 0; m < 4; ++m) {
                Frag af;
                const unsigned short* ap =
                    As + (((m * 16 + l16) << 8) + (kc << 5) + half * 8);
                af.u[0] = *(const uv4*)(ap);
                af.u[1] = *(const uv4*)(ap + 16);
                acc[m] = __builtin_amdgcn_wmma_f32_16x16x32_bf16(
                    false, af.v, false, bf.v, (short)0, acc[m], false, false);
            }
        }
    }

    if (mode == 0) {
        const float bv = bias ? bias[col] : 0.0f;
#pragma unroll
        for (int m = 0; m < 4; ++m) {
#pragma unroll
            for (int v = 0; v < 8; ++v) {
                int row = rowbase + m * 16 + v + half * 8;
                C[(size_t)row * N + col] = acc[m][v] + bv;
            }
        }
    } else {
        const float ws = wscore[col];
#pragma unroll
        for (int m = 0; m < 4; ++m) {
#pragma unroll
            for (int v = 0; v < 8; ++v) {
                int rl  = m * 16 + v + half * 8;
                int row = rowbase + rl;
                float t = tanhf(acc[m][v] + hproj[(row >> 8) * H_ + col]) * ws;
                t += __shfl_xor(t, 1, 16);
                t += __shfl_xor(t, 2, 16);
                t += __shfl_xor(t, 4, 16);
                t += __shfl_xor(t, 8, 16);
                if (l16 == 0) eacc[w * 64 + rl] = t;
            }
        }
        __syncthreads();
        if (tid < 64) {
            float s = eacc[tid] + eacc[64 + tid] + eacc[128 + tid] + eacc[192 + tid];
            epart[(size_t)blockIdx.y * M + rowbase + tid] = s;
        }
    }
}

// ---------------------------------------------------------------------------
// Kernel 3: softmax over T per batch. e = sum of npart partials.
// ---------------------------------------------------------------------------
__global__ void softmax_k(const float* __restrict__ epart, int M, int npart,
                          float* __restrict__ alpha, float* __restrict__ outAlpha)
{
    __shared__ float sm[256];
    int b = blockIdx.x, t = threadIdx.x;
    int row = b * T_ + t;
    float e = 0.0f;
    for (int p = 0; p < npart; ++p) e += epart[(size_t)p * M + row];
    sm[t] = e;
    __syncthreads();
    for (int s = 128; s > 0; s >>= 1) {
        if (t < s) sm[t] = fmaxf(sm[t], sm[t + s]);
        __syncthreads();
    }
    float mx = sm[0];
    __syncthreads();
    float ex = expf(e - mx);
    sm[t] = ex;
    __syncthreads();
    for (int s = 128; s > 0; s >>= 1) {
        if (t < s) sm[t] += sm[t + s];
        __syncthreads();
    }
    float a = ex / sm[0];
    alpha[row] = a;
    outAlpha[row] = a;
}

// ---------------------------------------------------------------------------
// Kernel 4: context = alpha^T batch_H, fused with building
//           x_full[b] = [context(512) | onehots(96) | prev_h(512)]
// ---------------------------------------------------------------------------
__global__ void context_k(const float* __restrict__ alpha, const float* __restrict__ batch_H,
                          const float* __restrict__ onehots, const float* __restrict__ prev_h,
                          float* __restrict__ xfull)
{
    __shared__ float al[T_];
    int b = blockIdx.x, t = threadIdx.x;
    al[t] = alpha[b * T_ + t];
    __syncthreads();
    float a0 = 0.0f, a1 = 0.0f;
    const float* bh = batch_H + (size_t)b * T_ * DE_;
    for (int tt = 0; tt < T_; ++tt) {
        float a = al[tt];
        const float* r = bh + (size_t)tt * DE_;
        a0 = fmaf(a, r[t], a0);
        a1 = fmaf(a, r[t + 256], a1);
    }
    float* x = xfull + (size_t)b * KX_;
    x[t] = a0;
    x[t + 256] = a1;
    if (t < NC_) x[DE_ + t] = onehots[b * NC_ + t];
    x[DE_ + NC_ + t]       = prev_h[b * H_ + t];
    x[DE_ + NC_ + 256 + t] = prev_h[b * H_ + 256 + t];
}

// ---------------------------------------------------------------------------
// Kernel 5: LSTM gates.  z[B,4H] -> h_new (out[0..]), c_new (out[B*H..])
// ---------------------------------------------------------------------------
__global__ void gates_k(const float* __restrict__ z, const float* __restrict__ prev_c,
                        float* __restrict__ out)
{
    int idx = blockIdx.x * 256 + threadIdx.x;   // 0 .. B*H-1
    int b = idx >> 9, h = idx & (H_ - 1);
    const float* zr = z + (size_t)b * N4H_;
    float gi = zr[h];
    float gf = zr[h + H_];
    float gg = zr[h + 2 * H_];
    float go = zr[h + 3 * H_];
    float c  = prev_c[idx];
    float cn = sigmoidf_(gf) * c + sigmoidf_(gi) * tanhf(gg);
    float hn = sigmoidf_(go) * tanhf(cn);
    out[idx] = hn;
    out[B_ * H_ + idx] = cn;
}

// ---------------------------------------------------------------------------
// Host launcher
// ---------------------------------------------------------------------------
extern "C" void kernel_launch(void* const* d_in, const int* in_sizes, int n_in,
                              void* d_out, int out_size, void* d_ws, size_t ws_size,
                              hipStream_t stream)
{
    (void)in_sizes; (void)n_in; (void)out_size; (void)ws_size;

    const float* prev_h   = (const float*)d_in[0];
    const float* prev_c   = (const float*)d_in[1];
    const float* batch_H  = (const float*)d_in[2];
    const float* onehots  = (const float*)d_in[3];
    const float* Wi2h     = (const float*)d_in[4];
    const float* Wh2h     = (const float*)d_in[5];
    const float* bh2h     = (const float*)d_in[6];
    const float* Wscore   = (const float*)d_in[7];
    const float* Wk       = (const float*)d_in[8];
    const float* Uk       = (const float*)d_in[9];
    const float* bk       = (const float*)d_in[10];
    float* out            = (float*)d_out;

    // ---- workspace layout (bytes) ----
    char* ws = (char*)d_ws;
    size_t o = 0;
    auto take = [&](size_t bytes) { size_t r = o; o = (o + bytes + 255) & ~(size_t)255; return r; };
    unsigned int* wsWi   = (unsigned int*)(ws + take((size_t)(H_/16) * (DE_/32) * 256 * 4));   // 512 KB
    unsigned int* wsWh   = (unsigned int*)(ws + take((size_t)(H_/16) * (H_/32)  * 256 * 4));   // 512 KB
    unsigned int* wsWkUk = (unsigned int*)(ws + take((size_t)(N4H_/16) * (KX_/32) * 256 * 4)); // 4.6 MB
    float* hprojW = (float*)(ws + take((size_t)B_ * H_ * 4));          // 512 KB
    float* epart  = (float*)(ws + take((size_t)8 * B_ * T_ * 4));      // 2 MB
    float* alphaW = (float*)(ws + take((size_t)B_ * T_ * 4));          // 256 KB
    float* xfull  = (float*)(ws + take((size_t)B_ * KX_ * 4));         // 1.1 MB
    float* zbuf   = (float*)(ws + take((size_t)B_ * N4H_ * 4));        // 2 MB

    // ---- 1. swizzle weights into WMMA bf16 B-fragments ----
    {   // Wi2h: K=512, N=512  -> 32 nb * 16 kc * 256 dwords
        int tot = (H_/16) * (DE_/32) * 256;
        swizzle_b_k<<<(tot + 255) / 256, 256, 0, stream>>>(Wi2h, wsWi, H_, DE_/32, 0, DE_/32);
    }
    {   // Wh2h: K=512, N=512
        int tot = (H_/16) * (H_/32) * 256;
        swizzle_b_k<<<(tot + 255) / 256, 256, 0, stream>>>(Wh2h, wsWh, H_, H_/32, 0, H_/32);
    }
    {   // Wk: K=608 (19 chunks), N=2048, kcBase=0, KCtot=35
        int tot = (N4H_/16) * 19 * 256;
        swizzle_b_k<<<(tot + 255) / 256, 256, 0, stream>>>(Wk, wsWkUk, N4H_, 19, 0, KX_/32);
    }
    {   // Uk: K=512 (16 chunks), N=2048, kcBase=19, KCtot=35
        int tot = (N4H_/16) * 16 * 256;
        swizzle_b_k<<<(tot + 255) / 256, 256, 0, stream>>>(Uk, wsWkUk, N4H_, 16, 19, KX_/32);
    }

    // ---- 2. h_proj = prev_h @ Wh2h + bh2h  (M=256,K=512,N=512) ----
    gemm_wmma_k<<<dim3(B_/64, H_/64), 128, 0, stream>>>(
        prev_h, wsWh, hprojW, bh2h, B_, H_, H_, H_/32, 0, nullptr, nullptr, nullptr);

    // ---- 3. attention scores: e partials (M=65536,K=512,N=512, fused tanh/score) ----
    gemm_wmma_k<<<dim3((B_*T_)/64, H_/64), 128, 0, stream>>>(
        batch_H, wsWi, nullptr, nullptr, B_*T_, H_, DE_, DE_/32, 1, hprojW, Wscore, epart);

    // ---- 4. softmax over T -> alpha (ws + out tail) ----
    softmax_k<<<B_, T_, 0, stream>>>(epart, B_*T_, H_/64, alphaW, out + 2 * B_ * H_);

    // ---- 5. context + build x_full = [context | onehots | prev_h] ----
    context_k<<<B_, 256, 0, stream>>>(alphaW, batch_H, onehots, prev_h, xfull);

    // ---- 6. z = x_full @ [Wk;Uk] + bk  (M=256,K=1120,N=2048) ----
    gemm_wmma_k<<<dim3(B_/64, N4H_/64), 128, 0, stream>>>(
        xfull, wsWkUk, zbuf, bk, B_, N4H_, KX_, KX_/32, 0, nullptr, nullptr, nullptr);

    // ---- 7. LSTM gates -> h_new, c_new ----
    gates_k<<<(B_ * H_) / 256, 256, 0, stream>>>(zbuf, prev_c, out);
}
